// RFDETRDecoder_1151051235434
// MI455X (gfx1250) — matrix operand order repaired and checked by reading, not compiled, not run on hardware
//
#include <hip/hip_runtime.h>
#include <math.h>

typedef __attribute__((ext_vector_type(16))) _Float16 v16h;
typedef __attribute__((ext_vector_type(8)))  float    v8f;

#define DMODEL 256
#define HEADS  8
#define HD     32
#define LVL    4
#define PTS    4
#define DFF    2048
#define BATCH  4
#define NQ     900
#define STOT   21760
#define MQ     (BATCH*NQ)   /* 3600 */

static __device__ __forceinline__ v8f wmma_f16(v16h a, v16h b, v8f c) {
  // D(16x16,f32) = A(16x32,f16) x B(32x16,f16) + C
  return __builtin_amdgcn_wmma_f32_16x16x32_f16(false, a, false, b, (short)0, c, false, false);
}

// ---------------------------------------------------------------------------
// Tensor Data Mover staging (gfx1250 TDM). Guarded: falls back to synchronous
// float4 staging if the builtin is unavailable on this toolchain.
// ---------------------------------------------------------------------------
#if defined(__has_builtin)
#if __has_builtin(__builtin_amdgcn_tensor_load_to_lds)
#define USE_TDM 1
#endif
#endif

#ifdef USE_TDM
typedef __attribute__((ext_vector_type(4))) unsigned int u32x4;
typedef __attribute__((ext_vector_type(8))) int          i32x8;
typedef __attribute__((ext_vector_type(4))) int          i32x4;

// 2D tile load, 4-byte elements: tile (tile0 x tile1), tensor dims for OOB
// zero-fill, row stride in elements. lds_addr = byte offset in LDS.
static __device__ __forceinline__ void tdm_load_2d(
    const void* gptr, unsigned lds_addr,
    unsigned tile0, unsigned tile1,
    unsigned dim0, unsigned dim1,
    unsigned long long stride0)
{
  unsigned long long ga = (unsigned long long)(uintptr_t)gptr;
  u32x4 g0;
  g0.x = 1u;                                               // count=1 (valid user D#)
  g0.y = lds_addr;                                         // bits 63:32  lds_addr
  g0.z = (unsigned)ga;                                     // bits 95:64  global_addr lo
  g0.w = (unsigned)((ga >> 32) & 0x1FFFFFFu) | (2u << 30); // gaddr[56:32] | type=2
  i32x8 g1;
  g1.s0 = (int)(2u << 16);                                  // data_size=4B (code 2)
  g1.s1 = (int)((dim0 & 0xFFFFu) << 16);                    // tensor_dim0[15:0]
  g1.s2 = (int)(((dim0 >> 16) & 0xFFFFu) | ((dim1 & 0xFFFFu) << 16));
  g1.s3 = (int)(((dim1 >> 16) & 0xFFFFu) | (tile0 << 16));  // tile_dim0
  g1.s4 = (int)(tile1 & 0xFFFFu);                           // tile_dim1, tile_dim2=0
  g1.s5 = (int)(unsigned)(stride0 & 0xFFFFFFFFull);         // tensor_dim0_stride lo
  g1.s6 = (int)(unsigned)((stride0 >> 32) & 0xFFFFull);     // stride hi | stride1=0
  g1.s7 = 0;
  i32x4 z4 = {0, 0, 0, 0};
#if __clang_major__ >= 23
  i32x8 z8 = {0, 0, 0, 0, 0, 0, 0, 0};
  __builtin_amdgcn_tensor_load_to_lds(g0, g1, z4, z4, z8, 0);
#else
  __builtin_amdgcn_tensor_load_to_lds(g0, g1, z4, z4, 0);
#endif
}

static __device__ __forceinline__ void wait_tensor0() {
#if __has_builtin(__builtin_amdgcn_s_wait_tensorcnt)
  __builtin_amdgcn_s_wait_tensorcnt(0);
#else
  asm volatile("s_wait_tensorcnt 0x0" ::: "memory");
#endif
}
#endif // USE_TDM

// ---------------------------------------------------------------------------
// GEMM: C[M,N] = A[M,K] @ W[K,N] + bias (+ReLU).
// Block tile 64x128, 8 waves in 2x4, each wave a 32x32 output (4 WMMA/k-step).
// K staged 32 at a time into LDS (f32, converted to f16 at fragment build).
// Requires N%128==0, K%32==0; M edge handled by OOB zero-fill / clamping.
// ---------------------------------------------------------------------------
__global__ __launch_bounds__(256) void gemm_kernel(
    const float* __restrict__ A, int lda,
    const float* __restrict__ W, int ldw,
    const float* __restrict__ bias,
    float* __restrict__ C, int ldc,
    int M, int K, int relu)
{
  __shared__ float lAf[64][32];    // [m][k]
  __shared__ float lBf[32][128];   // [k][n]
  const int tid   = threadIdx.x;
  const int lane  = tid & 31;
  const int wave  = tid >> 5;
  const int waveM = wave >> 2;     // 0..1  (32 rows each)
  const int waveN = wave & 3;      // 0..3  (32 cols each)
  const int m0 = blockIdx.y * 64;
  const int n0 = blockIdx.x * 128;
  const int nn = lane & 15;
  const int hh = lane >> 4;

  v8f acc00 = {}, acc01 = {}, acc10 = {}, acc11 = {};

  for (int k0 = 0; k0 < K; k0 += 32) {
#ifdef USE_TDM
    if (tid < 32) {
      unsigned mrem = (unsigned)(M - m0); if (mrem > 64u) mrem = 64u;
      // A tile: 64 rows x 32 k (row stride lda); OOB rows -> zero-filled LDS
      tdm_load_2d(A + (size_t)m0 * lda + k0,
                  (unsigned)(uintptr_t)&lAf[0][0],
                  32u, 64u, 32u, mrem, (unsigned long long)lda);
      // B tile: 32 k-rows x 128 n (row stride ldw)
      tdm_load_2d(W + (size_t)k0 * ldw + n0,
                  (unsigned)(uintptr_t)&lBf[0][0],
                  128u, 32u, 128u, 32u, (unsigned long long)ldw);
    }
    wait_tensor0();
    __syncthreads();
#else
    { // A tile: 8 f32 per thread
      int flat = tid * 8;
      int r = flat >> 5, c = flat & 31;
      int rr = m0 + r; if (rr > M - 1) rr = M - 1;
      const float4* src = (const float4*)(A + (size_t)rr * lda + k0 + c);
      *(float4*)&lAf[r][c]     = src[0];
      *(float4*)&lAf[r][c + 4] = src[1];
    }
    { // B tile: 16 f32 per thread, coalesced along n
      int flat = tid * 16;
      int kk = flat >> 7, nc = flat & 127;
      const float4* src = (const float4*)(W + (size_t)(k0 + kk) * ldw + n0 + nc);
      float4* dst = (float4*)&lBf[kk][nc];
      dst[0] = src[0]; dst[1] = src[1]; dst[2] = src[2]; dst[3] = src[3];
    }
    __syncthreads();
#endif
    // fragment build (f32 -> f16)
    v16h a0, a1, b0, b1;
    {
      int ra = waveM * 32 + nn;
      const float* p0 = &lAf[ra][hh * 16];
      const float* p1 = &lAf[ra + 16][hh * 16];
      #pragma unroll
      for (int j = 0; j < 8; ++j) {
        a0[j]     = (_Float16)p0[j];
        a0[8 + j] = (_Float16)p0[8 + j];
        a1[j]     = (_Float16)p1[j];
        a1[8 + j] = (_Float16)p1[8 + j];
      }
      int cn = waveN * 32 + nn;
      #pragma unroll
      for (int j = 0; j < 16; ++j) {
        b0[j] = (_Float16)lBf[hh * 16 + j][cn];
        b1[j] = (_Float16)lBf[hh * 16 + j][cn + 16];
      }
    }
    acc00 = wmma_f16(a0, b0, acc00);
    acc01 = wmma_f16(a0, b1, acc01);
    acc10 = wmma_f16(a1, b0, acc10);
    acc11 = wmma_f16(a1, b1, acc11);
    __syncthreads();
  }

  #pragma unroll
  for (int r = 0; r < 8; ++r) {
    int mrow0 = m0 + waveM * 32 + r + 8 * hh;   // C layout row striping
    int mrow1 = mrow0 + 16;
    int c0 = n0 + waveN * 32 + nn;
    int c1 = c0 + 16;
    float bi0 = bias[c0], bi1 = bias[c1];
    if (mrow0 < M) {
      float v0 = acc00[r] + bi0, v1 = acc01[r] + bi1;
      if (relu) { v0 = fmaxf(v0, 0.f); v1 = fmaxf(v1, 0.f); }
      C[(size_t)mrow0 * ldc + c0] = v0;
      C[(size_t)mrow0 * ldc + c1] = v1;
    }
    if (mrow1 < M) {
      float v0 = acc10[r] + bi0, v1 = acc11[r] + bi1;
      if (relu) { v0 = fmaxf(v0, 0.f); v1 = fmaxf(v1, 0.f); }
      C[(size_t)mrow1 * ldc + c0] = v0;
      C[(size_t)mrow1 * ldc + c1] = v1;
    }
  }
}

// ---------------------------------------------------------------------------
// Flash-style self-attention. One wave per (b, head, 16-query tile).
// Q/K/V are [B*NQ, 256] with head h at columns h*32..h*32+31.
// ---------------------------------------------------------------------------
__global__ __launch_bounds__(128) void attn_kernel(
    const float* __restrict__ Q, const float* __restrict__ Kh,
    const float* __restrict__ V, float* __restrict__ O)
{
  __shared__ _Float16 sP[4][16][32];
  const int tid  = threadIdx.x;
  const int lane = tid & 31;
  const int w    = tid >> 5;
  const int QT   = (NQ + 15) / 16;       // 57
  const int g    = blockIdx.x * 4 + w;
  const int qt = g % QT;
  const int h  = (g / QT) % HEADS;
  const int b  = g / (QT * HEADS);
  const int q0 = qt * 16;
  const int nn = lane & 15;
  const int hh = lane >> 4;
  const float scale = 0.17677669529663689f;  // HD^-0.5

  v16h aq;  // Q tile, pre-scaled, A-operand layout
  {
    int qr = q0 + nn; if (qr > NQ-1) qr = NQ-1;
    const float* qp = Q + ((size_t)(b*NQ + qr))*DMODEL + h*HD;
    #pragma unroll
    for (int i = 0; i < 8; ++i) {
      int ka = (i>>2)*16 + hh*8 + (i&3)*2;
      aq[2*i]   = (_Float16)(qp[ka]   * scale);
      aq[2*i+1] = (_Float16)(qp[ka+1] * scale);
    }
  }
  float mrow[8], lsum[8];
  #pragma unroll
  for (int r = 0; r < 8; ++r) { mrow[r] = -3.0e38f; lsum[r] = 0.f; }
  v8f o0 = {}, o1 = {};

  for (int kc = 0; kc < NQ; kc += 32) {
    v16h bk0, bk1;
    int k0i = kc + nn;       bool vk0 = k0i < NQ; int k0c = vk0 ? k0i : NQ-1;
    int k1i = kc + 16 + nn;  bool vk1 = k1i < NQ; int k1c = vk1 ? k1i : NQ-1;
    {
      const float* kp0 = Kh + ((size_t)(b*NQ + k0c))*DMODEL + h*HD;
      const float* kp1 = Kh + ((size_t)(b*NQ + k1c))*DMODEL + h*HD;
      #pragma unroll
      for (int i = 0; i < 8; ++i) {
        int kb = (i>>2)*16 + hh*8 + (i&3)*2;
        bk0[2*i] = (_Float16)kp0[kb]; bk0[2*i+1] = (_Float16)kp0[kb+1];
        bk1[2*i] = (_Float16)kp1[kb]; bk1[2*i+1] = (_Float16)kp1[kb+1];
      }
    }
    v8f z = {};
    v8f s0 = wmma_f16(aq, bk0, z);
    v8f s1 = wmma_f16(aq, bk1, z);
    #pragma unroll
    for (int r = 0; r < 8; ++r) {
      if (!vk0) s0[r] = -1e30f;
      if (!vk1) s1[r] = -1e30f;
    }
    float p0[8], p1[8];
    #pragma unroll
    for (int r = 0; r < 8; ++r) {
      float t = fmaxf(s0[r], s1[r]);
      t = fmaxf(t, __shfl_xor(t, 8, 32));
      t = fmaxf(t, __shfl_xor(t, 4, 32));
      t = fmaxf(t, __shfl_xor(t, 2, 32));
      t = fmaxf(t, __shfl_xor(t, 1, 32));
      float mn    = fmaxf(mrow[r], t);
      float alpha = __expf(mrow[r] - mn);
      p0[r] = __expf(s0[r] - mn);
      p1[r] = __expf(s1[r] - mn);
      float rs = p0[r] + p1[r];
      rs += __shfl_xor(rs, 8, 32);
      rs += __shfl_xor(rs, 4, 32);
      rs += __shfl_xor(rs, 2, 32);
      rs += __shfl_xor(rs, 1, 32);
      lsum[r] = lsum[r]*alpha + rs;
      o0[r] *= alpha; o1[r] *= alpha;
      mrow[r] = mn;
    }
    #pragma unroll
    for (int r = 0; r < 8; ++r) {
      int m = r + 8*hh;
      sP[w][m][nn]      = (_Float16)p0[r];
      sP[w][m][16 + nn] = (_Float16)p1[r];
    }
    __syncthreads();
    v16h ap;
    #pragma unroll
    for (int i = 0; i < 8; ++i) {
      int ka = (i>>2)*16 + hh*8 + (i&3)*2;
      ap[2*i]   = sP[w][nn][ka];
      ap[2*i+1] = sP[w][nn][ka+1];
    }
    v16h bv0, bv1;
    #pragma unroll
    for (int i = 0; i < 8; ++i) {
      int kk0 = kc + (i>>2)*16 + hh*8 + (i&3)*2;
      int kk1 = kk0 + 1;
      if (kk0 > NQ-1) kk0 = NQ-1;
      if (kk1 > NQ-1) kk1 = NQ-1;
      const float* vp0 = V + ((size_t)(b*NQ + kk0))*DMODEL + h*HD;
      const float* vp1 = V + ((size_t)(b*NQ + kk1))*DMODEL + h*HD;
      bv0[2*i]   = (_Float16)vp0[nn];
      bv0[2*i+1] = (_Float16)vp1[nn];
      bv1[2*i]   = (_Float16)vp0[16 + nn];
      bv1[2*i+1] = (_Float16)vp1[16 + nn];
    }
    o0 = wmma_f16(ap, bv0, o0);
    o1 = wmma_f16(ap, bv1, o1);
    __syncthreads();
  }
  #pragma unroll
  for (int r = 0; r < 8; ++r) {
    int m  = r + 8*hh;
    int qr = q0 + m;
    if (qr < NQ) {
      float inv = 1.0f / lsum[r];
      float* op = O + ((size_t)(b*NQ + qr))*DMODEL + h*HD;
      op[nn]      = o0[r]*inv;
      op[16 + nn] = o1[r]*inv;
    }
  }
}

// ---------------------------------------------------------------------------
__global__ void add_kernel(const float* __restrict__ a, const float* __restrict__ b,
                           float* __restrict__ o, int n) {
  int i = blockIdx.x * 256 + threadIdx.x;
  if (i < n) o[i] = a[i] + b[i];
}

// out = LayerNorm(a + b); one wave per row of 256
__global__ __launch_bounds__(128) void add_ln_kernel(
    const float* __restrict__ A, const float* __restrict__ Bv,
    const float* __restrict__ gam, const float* __restrict__ bet,
    float* __restrict__ out, int rows)
{
  int tid = threadIdx.x;
  int lane = tid & 31;
  int w = tid >> 5;
  int row = blockIdx.x * 4 + w;
  if (row >= rows) return;
  const float* pa = A  + (size_t)row * DMODEL;
  const float* pb = Bv + (size_t)row * DMODEL;
  float v[8];
  float s = 0.f;
  #pragma unroll
  for (int i = 0; i < 8; ++i) { v[i] = pa[lane*8+i] + pb[lane*8+i]; s += v[i]; }
  #pragma unroll
  for (int off = 16; off > 0; off >>= 1) s += __shfl_xor(s, off, 32);
  float mean = s * (1.0f/DMODEL);
  float vs = 0.f;
  #pragma unroll
  for (int i = 0; i < 8; ++i) { float d = v[i]-mean; vs += d*d; }
  #pragma unroll
  for (int off = 16; off > 0; off >>= 1) vs += __shfl_xor(vs, off, 32);
  float rstd = rsqrtf(vs * (1.0f/DMODEL) + 1e-5f);
  float* po = out + (size_t)row * DMODEL;
  #pragma unroll
  for (int i = 0; i < 8; ++i) {
    int d = lane*8+i;
    po[d] = (v[i]-mean)*rstd*gam[d] + bet[d];
  }
}

// softmax over each group of L*P=16 attention weights, in place
__global__ void softmax16_kernel(float* __restrict__ aw, int total) {
  int i = blockIdx.x * 256 + threadIdx.x;
  if (i >= total) return;
  int row = i / HEADS, h = i % HEADS;
  float* p = aw + (size_t)row * (HEADS*LVL*PTS) + h * (LVL*PTS);
  float v[16], mx = -3e38f;
  #pragma unroll
  for (int j = 0; j < 16; ++j) { v[j] = p[j]; mx = fmaxf(mx, v[j]); }
  float s = 0.f;
  #pragma unroll
  for (int j = 0; j < 16; ++j) { v[j] = __expf(v[j]-mx); s += v[j]; }
  float inv = 1.f / s;
  #pragma unroll
  for (int j = 0; j < 16; ++j) p[j] = v[j]*inv;
}

// Deformable bilinear sampling: one wave per (b,q,head), lane = head channel.
__global__ __launch_bounds__(128) void deform_kernel(
    const float* __restrict__ V, const float* __restrict__ off,
    const float* __restrict__ aw, const float* __restrict__ ref,
    const int* __restrict__ shapes, const int* __restrict__ starts,
    float* __restrict__ out)
{
  int tid  = threadIdx.x;
  int lane = tid & 31;
  int w    = tid >> 5;
  int g = blockIdx.x * 4 + w;
  int h = g % HEADS;
  int q = (g / HEADS) % NQ;
  int b = g / (HEADS * NQ);
  const float* offp = off + ((size_t)(b*NQ+q))*DMODEL + h*(LVL*PTS*2);
  const float* awp  = aw  + ((size_t)(b*NQ+q))*(HEADS*LVL*PTS) + h*(LVL*PTS);
  float acc = 0.f;
  for (int l = 0; l < LVL; ++l) {
    int Hs = shapes[l*2+0], Ws = shapes[l*2+1];
    int st = starts[l];
    const float* rp = ref + (((size_t)(b*NQ+q))*LVL + l)*4;
    float cx = rp[0], cy = rp[1], rw = rp[2], rh = rp[3];
    #pragma unroll
    for (int p = 0; p < PTS; ++p) {
      float ox = offp[l*(PTS*2) + p*2 + 0];
      float oy = offp[l*(PTS*2) + p*2 + 1];
      float a  = awp[l*PTS + p];
      float x = (cx + ox * (1.0f/PTS) * rw * 0.5f) * Ws - 0.5f;
      float y = (cy + oy * (1.0f/PTS) * rh * 0.5f) * Hs - 0.5f;
      float x0 = floorf(x), y0 = floorf(y);
      float dx = x - x0, dy = y - y0;
      int xi0 = (int)x0, yi0 = (int)y0;
      float cw[4] = { (1-dx)*(1-dy), dx*(1-dy), (1-dx)*dy, dx*dy };
      int xs[4] = { xi0, xi0+1, xi0,   xi0+1 };
      int ys[4] = { yi0, yi0,   yi0+1, yi0+1 };
      #pragma unroll
      for (int c = 0; c < 4; ++c) {
        int xi = xs[c], yi = ys[c];
        bool valid = (xi >= 0) && (xi < Ws) && (yi >= 0) && (yi < Hs);
        int xc = xi < 0 ? 0 : (xi > Ws-1 ? Ws-1 : xi);
        int yc = yi < 0 ? 0 : (yi > Hs-1 ? Hs-1 : yi);
        int idx = yc*Ws + xc;
        float val = V[((size_t)(b*STOT) + st + idx)*DMODEL + h*HD + lane];
        acc += a * cw[c] * (valid ? val : 0.f);
      }
    }
  }
  out[((size_t)(b*NQ+q))*DMODEL + h*HD + lane] = acc;
}

// ---------------------------------------------------------------------------
extern "C" void kernel_launch(void* const* d_in, const int* in_sizes, int n_in,
                              void* d_out, int out_size, void* d_ws, size_t ws_size,
                              hipStream_t stream) {
  (void)in_sizes; (void)n_in; (void)out_size; (void)ws_size;
  const float* query    = (const float*)d_in[0];
  const float* qpos     = (const float*)d_in[1];
  const float* refpts   = (const float*)d_in[2];
  const float* inflat   = (const float*)d_in[3];
  const int*   shapes   = (const int*)d_in[4];
  const int*   starts   = (const int*)d_in[5];
  const float* sa_in_w  = (const float*)d_in[6];
  const float* sa_in_b  = (const float*)d_in[7];
  const float* sa_out_w = (const float*)d_in[8];
  const float* sa_out_b = (const float*)d_in[9];
  const float* n1g = (const float*)d_in[10];
  const float* n1b = (const float*)d_in[11];
  const float* value_w  = (const float*)d_in[12];
  const float* value_b  = (const float*)d_in[13];
  const float* off_w    = (const float*)d_in[14];
  const float* off_b    = (const float*)d_in[15];
  const float* aw_w     = (const float*)d_in[16];
  const float* aw_b     = (const float*)d_in[17];
  const float* ca_out_w = (const float*)d_in[18];
  const float* ca_out_b = (const float*)d_in[19];
  const float* n2g = (const float*)d_in[20];
  const float* n2b = (const float*)d_in[21];
  const float* l1w = (const float*)d_in[22];
  const float* l1b = (const float*)d_in[23];
  const float* l2w = (const float*)d_in[24];
  const float* l2b = (const float*)d_in[25];
  const float* n3g = (const float*)d_in[26];
  const float* n3b = (const float*)d_in[27];
  float* out = (float*)d_out;

  float* ws = (float*)d_ws;
  const size_t SZ = (size_t)MQ * DMODEL;   // 921600
  float* slotE = ws;            // qk -> sa_heads -> ca_pre
  float* slot1 = ws + SZ;       // qh -> sa -> off -> ca -> ffn
  float* slot2 = ws + 2*SZ;     // kh -> x1p
  float* slot3 = ws + 3*SZ;     // vh -> aw -> x2
  float* slot4 = ws + 4*SZ;     // x1
  float* slotV = ws + 5*SZ;     // value [87040,256] -> FFN hidden [3600,2048]

  dim3 blk256(256), blk128(128);
  dim3 gP(DMODEL/128, (MQ + 63)/64);           // [3600,256]
  dim3 gA(1, (MQ + 63)/64);                    // [3600,128]
  dim3 gV(DMODEL/128, (BATCH*STOT)/64);        // [87040,256]
  dim3 gF1(DFF/128, (MQ + 63)/64);             // [3600,2048]

  // 1. qk = query + query_pos
  add_kernel<<<(MQ*DMODEL+255)/256, blk256, 0, stream>>>(query, qpos, slotE, MQ*DMODEL);
  // 2-4. QKV projections
  gemm_kernel<<<gP, blk256, 0, stream>>>(slotE, DMODEL, sa_in_w + 0*DMODEL*DMODEL, DMODEL, sa_in_b + 0*DMODEL, slot1, DMODEL, MQ, DMODEL, 0);
  gemm_kernel<<<gP, blk256, 0, stream>>>(slotE, DMODEL, sa_in_w + 1*DMODEL*DMODEL, DMODEL, sa_in_b + 1*DMODEL, slot2, DMODEL, MQ, DMODEL, 0);
  gemm_kernel<<<gP, blk256, 0, stream>>>(query, DMODEL, sa_in_w + 2*DMODEL*DMODEL, DMODEL, sa_in_b + 2*DMODEL, slot3, DMODEL, MQ, DMODEL, 0);
  // 5. self-attention -> slotE
  attn_kernel<<<(BATCH*HEADS*((NQ+15)/16))/4, blk128, 0, stream>>>(slot1, slot2, slot3, slotE);
  // 6. sa output projection
  gemm_kernel<<<gP, blk256, 0, stream>>>(slotE, DMODEL, sa_out_w, DMODEL, sa_out_b, slot1, DMODEL, MQ, DMODEL, 0);
  // 7. x1 = LN(query + sa)
  add_ln_kernel<<<(MQ+3)/4, blk128, 0, stream>>>(query, slot1, n1g, n1b, slot4, MQ);
  // 8. x1p = x1 + query_pos
  add_kernel<<<(MQ*DMODEL+255)/256, blk256, 0, stream>>>(slot4, qpos, slot2, MQ*DMODEL);
  // 9. value projection (big one)
  gemm_kernel<<<gV, blk256, 0, stream>>>(inflat, DMODEL, value_w, DMODEL, value_b, slotV, DMODEL, BATCH*STOT, DMODEL, 0);
  // 10. sampling offsets
  gemm_kernel<<<gP, blk256, 0, stream>>>(slot2, DMODEL, off_w, 256, off_b, slot1, 256, MQ, DMODEL, 0);
  // 11. attention-weight logits
  gemm_kernel<<<gA, blk256, 0, stream>>>(slot2, DMODEL, aw_w, 128, aw_b, slot3, 128, MQ, DMODEL, 0);
  // 12. softmax over 16 per (row, head)
  softmax16_kernel<<<(MQ*HEADS+255)/256, blk256, 0, stream>>>(slot3, MQ*HEADS);
  // 13. deformable sampling -> ca_pre
  deform_kernel<<<(BATCH*NQ*HEADS)/4, blk128, 0, stream>>>(slotV, slot1, slot3, refpts, shapes, starts, slotE);
  // 14. ca output projection
  gemm_kernel<<<gP, blk256, 0, stream>>>(slotE, DMODEL, ca_out_w, DMODEL, ca_out_b, slot1, DMODEL, MQ, DMODEL, 0);
  // 15. x2 = LN(x1 + ca)
  add_ln_kernel<<<(MQ+3)/4, blk128, 0, stream>>>(slot4, slot1, n2g, n2b, slot3, MQ);
  // 16. FFN up + ReLU (fused)
  gemm_kernel<<<gF1, blk256, 0, stream>>>(slot3, DMODEL, l1w, DFF, l1b, slotV, DFF, MQ, DMODEL, 1);
  // 17. FFN down
  gemm_kernel<<<gP, blk256, 0, stream>>>(slotV, DFF, l2w, DMODEL, l2b, slot1, DMODEL, MQ, DFF, 0);
  // 18. out = LN(x2 + ffn)
  add_ln_kernel<<<(MQ+3)/4, blk128, 0, stream>>>(slot3, slot1, n3g, n3b, out, MQ);
}
